// Attention_85882166051657
// MI455X (gfx1250) — matrix operand order repaired and checked by reading, not compiled, not run on hardware
//
#include <hip/hip_runtime.h>
#include <stdint.h>

// Problem constants (fixed by the reference)
#define BB   2
#define TT   2048
#define DDIM 1024
#define HH   16
#define DH   64

typedef __attribute__((ext_vector_type(16))) __bf16    v16bf;
typedef __attribute__((ext_vector_type(8)))  float     v8f;
typedef __attribute__((ext_vector_type(4)))  uint32_t  u32x4;

union FragAB { u32x4 q[2]; v16bf v; };

__device__ __forceinline__ unsigned short f32_to_bf16(float f) {
  union { float f; uint32_t u; } cv; cv.f = f;
  uint32_t u = cv.u;
  if ((u & 0x7f800000u) == 0x7f800000u) return (unsigned short)(u >> 16); // inf/nan passthrough
  return (unsigned short)((u + 0x7fffu + ((u >> 16) & 1u)) >> 16);        // RNE
}

// A-fragment (16-bit, 16x32): lane holds row (lane&15), K = {kb..kb+7, kb+16..kb+23}, kb = lane<16?0:8
__device__ __forceinline__ v16bf load_frag_a_g(const unsigned short* __restrict__ base,
                                               int ld, int row, int kbase, int lane) {
  const unsigned short* p = base + (size_t)row * ld + kbase + ((lane >> 4) << 3);
  FragAB f;
  f.q[0] = *(const u32x4*)(p);
  f.q[1] = *(const u32x4*)(p + 16);
  return f.v;
}

// B-fragment (16-bit, 32x16): lane holds col (lane&15), K = {kb..kb+7, kb+16..kb+23}, kb = lane<16?0:8
__device__ __forceinline__ v16bf load_frag_b_g(const unsigned short* __restrict__ base,
                                               int ld, int col, int kbase, int lane) {
  const unsigned short* p = base + (size_t)col * ld + kbase + ((lane >> 4) << 3);
  FragAB f;
  f.q[0] = *(const u32x4*)(p);
  f.q[1] = *(const u32x4*)(p + 16);
  return f.v;
}

#define WMMA_BF16(a, b, c) \
  __builtin_amdgcn_wmma_f32_16x16x32_bf16(false, (a), false, (b), (short)0, (c), false, false)

// ---------------- stage 1: conversions ----------------

__global__ __launch_bounds__(256) void cvt_f32_bf16_kernel(const float* __restrict__ in,
                                                           unsigned short* __restrict__ out, int n) {
  int i = blockIdx.x * 256 + threadIdx.x;
  if (i < n) out[i] = f32_to_bf16(in[i]);
}

// in: [D,D] f32 (k-major rows), out: [D,D] bf16 transposed (out[n][k] = in[k][n])
__global__ __launch_bounds__(256) void cvt_transpose_kernel(const float* __restrict__ in,
                                                            unsigned short* __restrict__ out) {
  __shared__ float tile[32][33];
  const int tx = threadIdx.x & 31;
  const int ty = threadIdx.x >> 5;           // 0..7
  const int k0 = blockIdx.x * 32;
  const int n0 = blockIdx.y * 32;
  #pragma unroll
  for (int i = ty; i < 32; i += 8)
    tile[i][tx] = in[(size_t)(k0 + i) * DDIM + n0 + tx];
  __syncthreads();
  #pragma unroll
  for (int i = ty; i < 32; i += 8)
    out[(size_t)(n0 + i) * DDIM + k0 + tx] = f32_to_bf16(tile[tx][i]);
}

// ---------------- stage 2/4: WMMA GEMM ----------------
// C[M,N] = A[M,K] @ Bt[N,K]^T + bias.
// Block = 128 threads (4 waves). Block tile: 128 rows x 64 cols.
// Each wave computes a 32x64 strip: 2 A-fragments x 4 B-fragments = 8 WMMAs per k-step,
// with a software-pipelined k-loop (next k fragments prefetched before the WMMAs issue).
// mode 0: Q  -> bf16 [B,H,T,DH], scaled          mode 1: K -> bf16 [B,H,T,DH]
// mode 2: V  -> bf16 [B,H,DH,T] (transposed)     mode 3: f32 d_out, * data_mask
__global__ __launch_bounds__(128) void gemm_bf16_kernel(
    const unsigned short* __restrict__ A,
    const unsigned short* __restrict__ Bt,
    const float* __restrict__ bias,
    const int* __restrict__ dmask,
    void* __restrict__ outp,
    int K, int mode, float scale)
{
  const int lane = threadIdx.x & 31;
  const int wave = threadIdx.x >> 5;
  const int nl   = lane & 15;
  const int m0 = blockIdx.x * 128 + wave * 32;
  const int n0 = blockIdx.y * 64;
  const int row0 = m0 + nl;
  const int row1 = row0 + 16;

  v8f acc[2][4] = {};

  // prologue: fragments for k = 0
  v16bf a0 = load_frag_a_g(A, K, row0, 0, lane);
  v16bf a1 = load_frag_a_g(A, K, row1, 0, lane);
  v16bf b0 = load_frag_b_g(Bt, K, n0 +  0 + nl, 0, lane);
  v16bf b1 = load_frag_b_g(Bt, K, n0 + 16 + nl, 0, lane);
  v16bf b2 = load_frag_b_g(Bt, K, n0 + 32 + nl, 0, lane);
  v16bf b3 = load_frag_b_g(Bt, K, n0 + 48 + nl, 0, lane);

  for (int k = 0; k < K; k += 32) {
    // prefetch next k-step before issuing WMMAs (hides load latency behind 8 WMMAs)
    v16bf na0 = a0, na1 = a1, nb0 = b0, nb1 = b1, nb2 = b2, nb3 = b3;
    const int kn = k + 32;
    if (kn < K) {
      na0 = load_frag_a_g(A, K, row0, kn, lane);
      na1 = load_frag_a_g(A, K, row1, kn, lane);
      nb0 = load_frag_b_g(Bt, K, n0 +  0 + nl, kn, lane);
      nb1 = load_frag_b_g(Bt, K, n0 + 16 + nl, kn, lane);
      nb2 = load_frag_b_g(Bt, K, n0 + 32 + nl, kn, lane);
      nb3 = load_frag_b_g(Bt, K, n0 + 48 + nl, kn, lane);
    }
    acc[0][0] = WMMA_BF16(a0, b0, acc[0][0]);
    acc[0][1] = WMMA_BF16(a0, b1, acc[0][1]);
    acc[0][2] = WMMA_BF16(a0, b2, acc[0][2]);
    acc[0][3] = WMMA_BF16(a0, b3, acc[0][3]);
    acc[1][0] = WMMA_BF16(a1, b0, acc[1][0]);
    acc[1][1] = WMMA_BF16(a1, b1, acc[1][1]);
    acc[1][2] = WMMA_BF16(a1, b2, acc[1][2]);
    acc[1][3] = WMMA_BF16(a1, b3, acc[1][3]);
    a0 = na0; a1 = na1; b0 = nb0; b1 = nb1; b2 = nb2; b3 = nb3;
  }

  const int mloc = (lane < 16) ? 0 : 8;   // C/D layout: half-wave holds rows M or M+8
  #pragma unroll
  for (int s = 0; s < 2; ++s) {
    #pragma unroll
    for (int t = 0; t < 4; ++t) {
      const int col = n0 + t * 16 + nl;
      const float bv = bias[col];
      #pragma unroll
      for (int r = 0; r < 8; ++r) {
        const int mm  = m0 + s * 16 + mloc + r;
        const float val = (acc[s][t][r] + bv) * scale;
        if (mode == 3) {
          ((float*)outp)[(size_t)mm * DDIM + col] = val * (float)dmask[mm];
        } else {
          const int b  = mm / TT, tok = mm % TT;
          const int h  = col / DH, dh  = col % DH;
          const unsigned short bfv = f32_to_bf16(val);
          if (mode == 2)
            ((unsigned short*)outp)[(((size_t)(b * HH + h)) * DH + dh) * TT + tok] = bfv;
          else
            ((unsigned short*)outp)[(((size_t)(b * HH + h)) * TT + tok) * DH + dh] = bfv;
        }
      }
    }
  }
}

// ---------------- stage 3: flash attention ----------------
// grid (B*H, T/64); 128 threads = 4 waves; each wave owns 16 query rows.
// Q pre-scaled by DH^-0.5. Keys streamed in 32-wide tiles, online softmax in f32.
// V fragments are loaded immediately after the S WMMAs so their latency hides
// under the softmax VALU/shuffle work.
__global__ __launch_bounds__(128) void flash_attn_kernel(
    const unsigned short* __restrict__ Q,    // [B,H,T,DH] bf16
    const unsigned short* __restrict__ Km,   // [B,H,T,DH] bf16
    const unsigned short* __restrict__ Vt,   // [B,H,DH,T] bf16
    const int* __restrict__ dmask,           // [B,T]
    unsigned short* __restrict__ Out)        // [B*T, D] bf16 (heads concatenated)
{
  __shared__ __align__(16) unsigned short Pld[4][16][40]; // 16x32 P tile / wave, padded rows (80B)

  const int lane  = threadIdx.x & 31;
  const int wave  = threadIdx.x >> 5;
  const int nl    = lane & 15;
  const int bh    = blockIdx.x;             // b*H + h
  const int b     = bh / HH, h = bh % HH;
  const int qtile = blockIdx.y;
  const int q0    = qtile * 64 + wave * 16;

  const unsigned short* Qb = Q  + (size_t)bh * TT * DH;
  const unsigned short* Kb = Km + (size_t)bh * TT * DH;
  const unsigned short* Vb = Vt + (size_t)bh * DH * TT;
  const int* dmb = dmask + b * TT;

  const int qrow = q0 + nl;
  const v16bf qa0 = load_frag_a_g(Qb, DH, qrow, 0, lane);   // DH k-steps 0..31
  const v16bf qa1 = load_frag_a_g(Qb, DH, qrow, 32, lane);  // 32..63

  const int   mloc = (lane < 16) ? 0 : 8;
  const float NEG  = -__builtin_inff();
  float mstat[8], lstat[8];
  int   dmq[8];
  v8f   acc[4] = {};
  #pragma unroll
  for (int r = 0; r < 8; ++r) {
    mstat[r] = NEG; lstat[r] = 0.f;
    dmq[r]   = dmb[q0 + mloc + r];
  }

  const int kend = qtile * 64 + 64;   // causal bound, uniform over the block (barrier-safe)
  for (int kt = 0; kt < kend; kt += 32) {
    // S = Q K^T, 16x32 tile as two 16x16 accumulators
    v8f s0 = {}, s1 = {};
    {
      v16bf kb0  = load_frag_b_g(Kb, DH, kt + nl,      0,  lane);
      v16bf kb0b = load_frag_b_g(Kb, DH, kt + nl,      32, lane);
      v16bf kb1  = load_frag_b_g(Kb, DH, kt + 16 + nl, 0,  lane);
      v16bf kb1b = load_frag_b_g(Kb, DH, kt + 16 + nl, 32, lane);
      s0 = WMMA_BF16(qa0, kb0,  s0);
      s0 = WMMA_BF16(qa1, kb0b, s0);
      s1 = WMMA_BF16(qa0, kb1,  s1);
      s1 = WMMA_BF16(qa1, kb1b, s1);
    }

    // V fragments for this key tile: independent of softmax, issue loads now
    v16bf vb0 = load_frag_b_g(Vb, TT,  0 + nl, kt, lane);
    v16bf vb1 = load_frag_b_g(Vb, TT, 16 + nl, kt, lane);
    v16bf vb2 = load_frag_b_g(Vb, TT, 32 + nl, kt, lane);
    v16bf vb3 = load_frag_b_g(Vb, TT, 48 + nl, kt, lane);

    const int key0 = kt + nl;
    const int key1 = key0 + 16;
    const int dk0  = dmb[key0];
    const int dk1  = dmb[key1];

    float e0[8], e1[8];
    #pragma unroll
    for (int r = 0; r < 8; ++r) {
      const int q = q0 + mloc + r;
      float v0 = s0[r], v1 = s1[r];
      if (key0 > q || !dk0 || !dmq[r]) v0 = NEG;
      if (key1 > q || !dk1 || !dmq[r]) v1 = NEG;
      float tmax = fmaxf(v0, v1);
      for (int off = 1; off < 16; off <<= 1) tmax = fmaxf(tmax, __shfl_xor(tmax, off, 16));
      const float nm = fmaxf(mstat[r], tmax);
      const float sc = (mstat[r] == nm) ? 1.0f : __expf(mstat[r] - nm);
      const float p0 = (v0 == NEG) ? 0.f : __expf(v0 - nm);
      const float p1 = (v1 == NEG) ? 0.f : __expf(v1 - nm);
      float psum = p0 + p1;
      for (int off = 1; off < 16; off <<= 1) psum += __shfl_xor(psum, off, 16);
      lstat[r] = lstat[r] * sc + psum;
      mstat[r] = nm;
      #pragma unroll
      for (int t = 0; t < 4; ++t) acc[t][r] *= sc;
      e0[r] = p0; e1[r] = p1;
    }

    // D-layout -> A-layout via LDS (bf16)
    __syncthreads();
    #pragma unroll
    for (int r = 0; r < 8; ++r) {
      Pld[wave][mloc + r][nl]      = f32_to_bf16(e0[r]);
      Pld[wave][mloc + r][nl + 16] = f32_to_bf16(e1[r]);
    }
    __syncthreads();

    v16bf pa;
    {
      const unsigned short* p = &Pld[wave][nl][(lane >> 4) << 3];
      FragAB f;
      f.q[0] = *(const u32x4*)(p);
      f.q[1] = *(const u32x4*)(p + 16);
      pa = f.v;
    }

    // O += P @ V  (V fragments already resident)
    acc[0] = WMMA_BF16(pa, vb0, acc[0]);
    acc[1] = WMMA_BF16(pa, vb1, acc[1]);
    acc[2] = WMMA_BF16(pa, vb2, acc[2]);
    acc[3] = WMMA_BF16(pa, vb3, acc[3]);
  }

  // epilogue: O /= l (fully-masked rows -> 0), write [B*T, D] bf16
  #pragma unroll
  for (int t = 0; t < 4; ++t) {
    #pragma unroll
    for (int r = 0; r < 8; ++r) {
      const int q = q0 + mloc + r;
      const float inv = (lstat[r] > 0.f) ? 1.0f / lstat[r] : 0.f;
      Out[((size_t)(b * TT + q)) * DDIM + h * DH + t * 16 + nl] =
          f32_to_bf16(acc[t][r] * inv);
    }
  }
}

// ---------------- launch ----------------

extern "C" void kernel_launch(void* const* d_in, const int* in_sizes, int n_in,
                              void* d_out, int out_size, void* d_ws, size_t ws_size,
                              hipStream_t stream) {
  const float* x  = (const float*)d_in[0];
  const int*   dm = (const int*)  d_in[1];
  const float* Wq = (const float*)d_in[2];
  const float* bq = (const float*)d_in[3];
  const float* Wk = (const float*)d_in[4];
  const float* bk = (const float*)d_in[5];
  const float* Wv = (const float*)d_in[6];
  const float* bv = (const float*)d_in[7];
  const float* Wp = (const float*)d_in[8];
  const float* bp = (const float*)d_in[9];
  float* out = (float*)d_out;

  // workspace layout (48 MiB total)
  char* ws = (char*)d_ws;
  const size_t MB = 1024u * 1024u;
  unsigned short* xbf = (unsigned short*)(ws + 0 * MB);   // [4096,1024] bf16 (8 MiB)
  unsigned short* Wqt = (unsigned short*)(ws + 8 * MB);   // [1024,1024] bf16 transposed (2 MiB each)
  unsigned short* Wkt = (unsigned short*)(ws + 10 * MB);
  unsigned short* Wvt = (unsigned short*)(ws + 12 * MB);
  unsigned short* Wpt = (unsigned short*)(ws + 14 * MB);
  unsigned short* Qb  = (unsigned short*)(ws + 16 * MB);  // [B,H,T,DH] bf16 (8 MiB)
  unsigned short* Kb  = (unsigned short*)(ws + 24 * MB);  // [B,H,T,DH]
  unsigned short* Vtb = (unsigned short*)(ws + 32 * MB);  // [B,H,DH,T]
  unsigned short* Ao  = (unsigned short*)(ws + 40 * MB);  // [B*T, D] bf16 attention out

  const int NX = BB * TT * DDIM;
  cvt_f32_bf16_kernel<<<(NX + 255) / 256, 256, 0, stream>>>(x, xbf, NX);

  dim3 tg(DDIM / 32, DDIM / 32);
  cvt_transpose_kernel<<<tg, 256, 0, stream>>>(Wq, Wqt);
  cvt_transpose_kernel<<<tg, 256, 0, stream>>>(Wk, Wkt);
  cvt_transpose_kernel<<<tg, 256, 0, stream>>>(Wv, Wvt);
  cvt_transpose_kernel<<<tg, 256, 0, stream>>>(Wp, Wpt);

  dim3 gg((BB * TT) / 128, DDIM / 64);
  gemm_bf16_kernel<<<gg, 128, 0, stream>>>(xbf, Wqt, bq, dm, Qb,  DDIM, 0, 0.125f); // DH^-0.5
  gemm_bf16_kernel<<<gg, 128, 0, stream>>>(xbf, Wkt, bk, dm, Kb,  DDIM, 1, 1.0f);
  gemm_bf16_kernel<<<gg, 128, 0, stream>>>(xbf, Wvt, bv, dm, Vtb, DDIM, 2, 1.0f);

  flash_attn_kernel<<<dim3(BB * HH, TT / 64), 128, 0, stream>>>(Qb, Kb, Vtb, dm, Ao);

  gemm_bf16_kernel<<<gg, 128, 0, stream>>>(Ao, Wpt, bp, dm, (void*)out, DDIM, 3, 1.0f);
}
